// MHA_88742614270715
// MI455X (gfx1250) — matrix operand order repaired and checked by reading, not compiled, not run on hardware
//
#include <hip/hip_runtime.h>

// MHA forward for MI455X (gfx1250, wave32, WMMA).
// Round 4: fused scores+softmax+PV kernel (16x2048 score block resident in
// 132 KB of the WGP's 320 KB LDS; `w` crosses HBM exactly once). Phase 1
// processes two 16x16 tiles per iteration (2 accumulators, batched fragment
// loads) and phase 3 is k-unrolled x2 so loads overlap WMMA and the
// WMMA->VALU hazard nops amortize over more work.

typedef __attribute__((ext_vector_type(16))) __bf16 v16bf;
typedef __attribute__((ext_vector_type(8)))  __bf16 bf16x8;
typedef __attribute__((ext_vector_type(8)))  float  v8f;

#define NUM_HEADS 16
#define D_MODEL   1024
#define DEPTH     64
#define BS        2
#define SEQ       2048
#define NTOK      (BS * SEQ)   // 4096
#define SC_PITCH  2052         // score row pitch in floats (pad 4 -> bank-shift)

union FragAB { v16bf v; bf16x8 h[2]; };

__device__ __forceinline__ __bf16 f2bf(float f) {
  return static_cast<__bf16>(f);   // native v_cvt (RNE) on gfx1250
}

// Load 8 contiguous fp32 (global or LDS) and convert to a bf16x8 half-fragment.
__device__ __forceinline__ bf16x8 cvt8(const float* p) {
  float4 a = *(const float4*)p;
  float4 b = *(const float4*)(p + 4);
  bf16x8 r;
  r[0] = f2bf(a.x); r[1] = f2bf(a.y); r[2] = f2bf(a.z); r[3] = f2bf(a.w);
  r[4] = f2bf(b.x); r[5] = f2bf(b.y); r[6] = f2bf(b.z); r[7] = f2bf(b.w);
  return r;
}

__device__ __forceinline__ v8f wmma_bf16(const FragAB& a, const FragAB& b, v8f c) {
  return __builtin_amdgcn_wmma_f32_16x16x32_bf16(false, a.v, false, b.v,
                                                 (short)0, c, false, false);
}

// B-operand fragment: lane holds one column; K 0..7|8..15 then 16..23|24..31.
__device__ __forceinline__ FragAB load_frag_b(const __bf16* base, int hl) {
  FragAB f;
  f.h[0] = *(const bf16x8*)(base + hl * 8);
  f.h[1] = *(const bf16x8*)(base + 16 + hl * 8);
  return f;
}

// A-operand fragment: lane holds one row; K 0..7|8..15 then 16..23|24..31.
__device__ __forceinline__ FragAB load_frag_a(const __bf16* base, int hl) {
  FragAB f;
  f.h[0] = *(const bf16x8*)(base + hl * 8);
  f.h[1] = *(const bf16x8*)(base + 16 + hl * 8);
  return f;
}

// ---------------------------------------------------------------------------
// Transpose-pack weights: Wt[n][k] = bf16(W[k][n]), 1024x1024.
// ---------------------------------------------------------------------------
__global__ void packT_kernel(const float* __restrict__ W, __bf16* __restrict__ Wt) {
  int k = blockIdx.x * 32 + threadIdx.x;
  int n = blockIdx.y * 8 + threadIdx.y;
  Wt[(size_t)n * D_MODEL + k] = f2bf(W[(size_t)k * D_MODEL + n]);
}

// ---------------------------------------------------------------------------
// Projection GEMM: Y = X(fp32,[4096x1024]) @ Wt^T(bf16, n-major) + bias.
// One wave -> 16x64 tile (4 accumulators, A reused 4x). Block (32,4).
// MODE 0: fp32 row-major out.  MODE 1: bf16 head-split [B,H,S,64].
// MODE 2: bf16 head-split transposed [B,H,64,S].
// ---------------------------------------------------------------------------
template <int MODE>
__global__ void proj_kernel(const float* __restrict__ A, const __bf16* __restrict__ Bt,
                            const float* __restrict__ bias, float* __restrict__ outF,
                            __bf16* __restrict__ outH, float scale) {
  const int lane = threadIdx.x;
  const int hl   = lane >> 4;
  const int l16  = lane & 15;
  const int ngrp  = blockIdx.x * 4 + threadIdx.y;   // 0..15, 64 columns each
  const int mtile = blockIdx.y;
  const int n0 = ngrp * 64;
  const float*  Arow = A  + (size_t)(mtile * 16 + l16) * D_MODEL;
  const __bf16* B0   = Bt + (size_t)(n0 + l16) * D_MODEL;
  v8f acc[4] = {};
  for (int k0 = 0; k0 < D_MODEL; k0 += 32) {
    __builtin_prefetch(Arow + k0 + 64, 0, 0);
    FragAB a;
    a.h[0] = cvt8(Arow + k0 + hl * 8);         // K 0..7 | 8..15
    a.h[1] = cvt8(Arow + k0 + 16 + hl * 8);    // K 16..23 | 24..31
#pragma unroll
    for (int j = 0; j < 4; ++j) {
      FragAB b = load_frag_b(B0 + (size_t)j * 16 * D_MODEL + k0, hl);
      acc[j] = wmma_bf16(a, b, acc[j]);
    }
  }
#pragma unroll
  for (int j = 0; j < 4; ++j) {
    const int n = n0 + j * 16 + l16;
    const float bn = bias[n];
#pragma unroll
    for (int r = 0; r < 8; ++r) {
      const int mrow = mtile * 16 + r + 8 * hl;  // D layout: vgpr r -> M = r + 8*half
      const float val = (acc[j][r] + bn) * scale;
      if (MODE == 0) {
        outF[(size_t)mrow * D_MODEL + n] = val;
      } else {
        const int bidx = mrow >> 11, s = mrow & (SEQ - 1);
        const int h = n >> 6, d = n & (DEPTH - 1);
        if (MODE == 1)
          outH[((size_t)(bidx * NUM_HEADS + h) * SEQ + s) * DEPTH + d] = f2bf(val);
        else
          outH[((size_t)(bidx * NUM_HEADS + h) * DEPTH + d) * SEQ + s] = f2bf(val);
      }
    }
  }
}

// ---------------------------------------------------------------------------
// Fused attention: per block = (bh, 16 q-rows).
//   Phase 1: scores (Q/8 . K + mask) -> LDS   [2 tiles/iter, 4 WMMA/iter]
//   Phase 2: softmax in LDS, stream normalized w to HBM once (float4)
//   Phase 3: out = w @ V^T, A-fragments from LDS  [k-unrolled x2]
// Block (32,4) = 4 waves; dynamic LDS = 16*SC_PITCH + reduce scratch.
// ---------------------------------------------------------------------------
__global__ void fused_attn_kernel(const __bf16* __restrict__ Qh,
                                  const __bf16* __restrict__ Kh,
                                  const __bf16* __restrict__ VhT,
                                  const float* __restrict__ mask,
                                  float* __restrict__ wout,
                                  float* __restrict__ attn) {
  extern __shared__ float smem[];
  float* sc   = smem;                   // [16][SC_PITCH]
  float* red  = smem + 16 * SC_PITCH;   // [8][16]
  float* rowv = red + 128;              // [32]: 0..15 row max, 16..31 row 1/sum
  const int lane = threadIdx.x, hl = lane >> 4, l16 = lane & 15;
  const int ty  = threadIdx.y;          // wave 0..3
  const int tid = ty * 32 + lane;
  const int q0 = blockIdx.x * 16;
  const int bh = blockIdx.y;
  const int b = bh >> 4, h = bh & 15;

  // ---- Phase 1: scores into LDS; wave ty covers k-positions [ty*512, +512)
  FragAB qa[2];
  {
    const __bf16* Qrow = Qh + ((size_t)bh * SEQ + q0 + l16) * DEPTH;
    qa[0] = load_frag_a(Qrow, hl);
    qa[1] = load_frag_a(Qrow + 32, hl);
  }
  for (int nt = 0; nt < 32; nt += 2) {
    const int n0 = ty * 512 + nt * 16;
    const __bf16* Krow0 = Kh + ((size_t)bh * SEQ + n0 + l16) * DEPTH;
    const __bf16* Krow1 = Krow0 + 16 * DEPTH;
    __builtin_prefetch(Krow0 + 32 * DEPTH, 0, 0);
    // Batch all 8 fragment loads, then 4 WMMAs on 2 independent accumulators.
    FragAB b00 = load_frag_b(Krow0,      hl);
    FragAB b01 = load_frag_b(Krow0 + 32, hl);
    FragAB b10 = load_frag_b(Krow1,      hl);
    FragAB b11 = load_frag_b(Krow1 + 32, hl);
    v8f a0 = {}, a1 = {};
    a0 = wmma_bf16(qa[0], b00, a0);
    a1 = wmma_bf16(qa[0], b10, a1);
    a0 = wmma_bf16(qa[1], b01, a0);
    a1 = wmma_bf16(qa[1], b11, a1);
    const int na = n0 + l16;
    const float mt0 = mask[(size_t)b * SEQ + na] * -1e9f;
    const float mt1 = mask[(size_t)b * SEQ + na + 16] * -1e9f;
#pragma unroll
    for (int r = 0; r < 8; ++r) {
      const int rr = (r + 8 * hl) * SC_PITCH;
      sc[rr + na]      = a0[r] + mt0;
      sc[rr + na + 16] = a1[r] + mt1;
    }
  }
  __syncthreads();

  // ---- Phase 2: softmax over each of 16 rows (2048 elems each)
  const int row = tid & 15, chunk = tid >> 4;          // 8 chunks x 256 floats
  float4* rp4 = (float4*)(sc + row * SC_PITCH + chunk * 256);
  float m = -3.4e38f;
  for (int i = 0; i < 64; ++i) {
    float4 t = rp4[i];
    m = fmaxf(m, fmaxf(fmaxf(t.x, t.y), fmaxf(t.z, t.w)));
  }
  red[chunk * 16 + row] = m;
  __syncthreads();
  if (tid < 16) {
    float mm = red[tid];
#pragma unroll
    for (int c = 1; c < 8; ++c) mm = fmaxf(mm, red[c * 16 + tid]);
    rowv[tid] = mm;
  }
  __syncthreads();
  const float rm = rowv[row];
  float s = 0.f;
  for (int i = 0; i < 64; ++i) {
    float4 t = rp4[i];
    t.x = __expf(t.x - rm); t.y = __expf(t.y - rm);
    t.z = __expf(t.z - rm); t.w = __expf(t.w - rm);
    s += (t.x + t.y) + (t.z + t.w);
    rp4[i] = t;
  }
  red[chunk * 16 + row] = s;
  __syncthreads();
  if (tid < 16) {
    float ss = red[tid];
#pragma unroll
    for (int c = 1; c < 8; ++c) ss += red[c * 16 + tid];
    rowv[16 + tid] = 1.0f / ss;
  }
  __syncthreads();

  // normalize in LDS + stream w to HBM exactly once (float4, coalesced)
  float* wbase = wout + ((size_t)bh * SEQ + q0) * SEQ;
  for (int i4 = tid; i4 < 16 * (SEQ / 4); i4 += 128) {
    const int r = i4 >> 9;               // 512 float4 per row
    const int c = (i4 & 511) * 4;
    float4 t = *(float4*)(sc + r * SC_PITCH + c);
    const float inv = rowv[16 + r];
    t.x *= inv; t.y *= inv; t.z *= inv; t.w *= inv;
    *(float4*)(sc + r * SC_PITCH + c) = t;
    *(float4*)(wbase + (size_t)r * SEQ + c) = t;
  }
  __syncthreads();

  // ---- Phase 3: out = w @ V^T; wave ty owns depth tile ty (16 of 64 cols)
  const __bf16* V0 = VhT + ((size_t)bh * DEPTH + ty * 16 + l16) * SEQ;
  const float* Asc = sc + l16 * SC_PITCH;
  v8f acc = {};
  for (int k0 = 0; k0 < SEQ; k0 += 64) {               // 2 WMMA per iteration
    __builtin_prefetch(V0 + k0 + 128, 0, 0);
    FragAB a0, a1;
    a0.h[0] = cvt8(Asc + k0 + hl * 8);
    a0.h[1] = cvt8(Asc + k0 + 16 + hl * 8);
    a1.h[0] = cvt8(Asc + k0 + 32 + hl * 8);
    a1.h[1] = cvt8(Asc + k0 + 48 + hl * 8);
    FragAB b0 = load_frag_b(V0 + k0,      hl);
    FragAB b1 = load_frag_b(V0 + k0 + 32, hl);
    acc = wmma_bf16(a0, b0, acc);
    acc = wmma_bf16(a1, b1, acc);
  }
  const int n = ty * 16 + l16;
#pragma unroll
  for (int r = 0; r < 8; ++r) {
    const int t = b * SEQ + q0 + r + 8 * hl;
    attn[(size_t)t * D_MODEL + h * DEPTH + n] = acc[r];
  }
}

// ---------------------------------------------------------------------------
extern "C" void kernel_launch(void* const* d_in, const int* in_sizes, int n_in,
                              void* d_out, int out_size, void* d_ws, size_t ws_size,
                              hipStream_t stream) {
  const float* q    = (const float*)d_in[0];
  const float* k    = (const float*)d_in[1];
  const float* v    = (const float*)d_in[2];
  const float* mask = (const float*)d_in[3];
  const float* Wq   = (const float*)d_in[4];
  const float* bq   = (const float*)d_in[5];
  const float* Wk   = (const float*)d_in[6];
  const float* bk   = (const float*)d_in[7];
  const float* Wv   = (const float*)d_in[8];
  const float* bv   = (const float*)d_in[9];
  const float* Wo   = (const float*)d_in[10];
  const float* bo   = (const float*)d_in[11];

  char* ws = (char*)d_ws;
  size_t off = 0;
  const size_t WBYTES = (size_t)D_MODEL * D_MODEL * sizeof(__bf16);  // 2 MB
  const size_t HBYTES = (size_t)NTOK * D_MODEL * sizeof(__bf16);     // 8 MB
  __bf16* WqT  = (__bf16*)(ws + off); off += WBYTES;
  __bf16* WkT  = (__bf16*)(ws + off); off += WBYTES;
  __bf16* WvT  = (__bf16*)(ws + off); off += WBYTES;
  __bf16* WoT  = (__bf16*)(ws + off); off += WBYTES;
  __bf16* Qh   = (__bf16*)(ws + off); off += HBYTES;   // [B,H,S,64]
  __bf16* Kh   = (__bf16*)(ws + off); off += HBYTES;   // [B,H,S,64]
  __bf16* VhT  = (__bf16*)(ws + off); off += HBYTES;   // [B,H,64,S]
  float*  attn = (float*)(ws + off);                   // [4096,1024] fp32

  float* out  = (float*)d_out;                                  // [2,2048,1024]
  float* wout = (float*)d_out + (size_t)NTOK * D_MODEL;         // [2,16,2048,2048]

  dim3 bT(32, 8), gT(D_MODEL / 32, D_MODEL / 8);
  packT_kernel<<<gT, bT, 0, stream>>>(Wq, WqT);
  packT_kernel<<<gT, bT, 0, stream>>>(Wk, WkT);
  packT_kernel<<<gT, bT, 0, stream>>>(Wv, WvT);
  packT_kernel<<<gT, bT, 0, stream>>>(Wo, WoT);

  dim3 bP(32, 4), gP(D_MODEL / 256, NTOK / 16);  // (4, 256)
  proj_kernel<1><<<gP, bP, 0, stream>>>(q, WqT, bq, nullptr, Qh,  0.125f); // fold 1/sqrt(64)
  proj_kernel<1><<<gP, bP, 0, stream>>>(k, WkT, bk, nullptr, Kh,  1.0f);
  proj_kernel<2><<<gP, bP, 0, stream>>>(v, WvT, bv, nullptr, VhT, 1.0f);

  const size_t smem_bytes = (size_t)(16 * SC_PITCH + 128 + 32) * sizeof(float); // ~132 KB
  dim3 gF(SEQ / 16, BS * NUM_HEADS);             // (128, 32)
  fused_attn_kernel<<<gF, bP, smem_bytes, stream>>>(Qh, Kh, VhT, mask, wout, attn);

  proj_kernel<0><<<gP, bP, 0, stream>>>(attn, WoT, bo, out, nullptr, 1.0f);
}
